// RBF_13589276525030
// MI455X (gfx1250) — matrix operand order repaired and checked by reading, not compiled, not run on hardware
//
#include <hip/hip_runtime.h>
#include <hip/hip_bf16.h>
#include <math.h>

typedef __attribute__((ext_vector_type(16))) _Float16 v16h;
typedef __attribute__((ext_vector_type(8)))  _Float16 v8h;
typedef __attribute__((ext_vector_type(2)))  _Float16 h2;
typedef __attribute__((ext_vector_type(8)))  float    v8f;

#define LOG2E 1.4426950408889634f

// ---------------------------------------------------------------------------
// Prep: one wave (32 lanes) per row of 64 floats.
//  - split each f32 into f16 hi + f16 lo residual (stored row-major)
//  - wave-reduce |row|^2, store b = -0.5 * |row|^2 * log2e / sigma^2
// ---------------------------------------------------------------------------
__global__ void rbf_prep(const float* __restrict__ src,
                         _Float16* __restrict__ hi,
                         _Float16* __restrict__ lo,
                         float* __restrict__ b,
                         const float* __restrict__ sigma,
                         int nrows) {
    const int wave = (int)((blockIdx.x * blockDim.x + threadIdx.x) >> 5);
    const int lane = (int)(threadIdx.x & 31);
    if (wave >= nrows) return;

    const float* r = src + (size_t)wave * 64;
    float v0 = r[2 * lane + 0];
    float v1 = r[2 * lane + 1];

    _Float16 h0 = (_Float16)v0, h1 = (_Float16)v1;
    _Float16 l0 = (_Float16)(v0 - (float)h0);
    _Float16 l1 = (_Float16)(v1 - (float)h1);

    h2 ph = {h0, h1};
    h2 pl = {l0, l1};
    *(h2*)(hi + (size_t)wave * 64 + 2 * lane) = ph;
    *(h2*)(lo + (size_t)wave * 64 + 2 * lane) = pl;

    float s = v0 * v0 + v1 * v1;
    #pragma unroll
    for (int off = 16; off > 0; off >>= 1)
        s += __shfl_xor(s, off, 32);

    if (lane == 0) {
        float sg = sigma[0];
        float k2 = LOG2E / (sg * sg);
        b[wave] = -0.5f * s * k2;
    }
}

// ---------------------------------------------------------------------------
// Main: each wave computes a 16x128 strip of the output.
// cross term via split-f16 WMMA: x.y ~= hi*hi + hi*lo + lo*hi  (fp32-quality)
// out = exp2( min( cross*k + bx[i] + by[j], 0 ) )   via native v_exp_f32
// ---------------------------------------------------------------------------
__global__ void __launch_bounds__(256)
rbf_main(const _Float16* __restrict__ xh, const _Float16* __restrict__ xl,
         const _Float16* __restrict__ yh, const _Float16* __restrict__ yl,
         const float* __restrict__ bx, const float* __restrict__ by,
         const float* __restrict__ sigma,
         float* __restrict__ out, int M) {
    const int lane = (int)(threadIdx.x & 31);
    const int wave = (int)(blockIdx.x * (blockDim.x >> 5) + (threadIdx.x >> 5));

    const int colBlocks = M >> 7;            // 128 output columns per wave
    const int rowTile   = wave / colBlocks;
    const int colBlock  = wave % colBlocks;
    const int i0 = rowTile << 4;
    const int j0 = colBlock << 7;

    const int hrow = lane & 15;              // row (A) / column (B,C,D) in tile
    const int hsel = lane >> 4;              // lane-half select

    union A16 { v16h v; v8h h[2]; };

    // --- A operands: rows i0+hrow of x (hi & lo), both K-steps, in registers.
    // A k order used here (contiguous, not AMD's table order): lane holds K = base..base+15,
    // base = hsel*16, +32 per K-step  -> two contiguous 16B chunks per operand.
    A16 ahi[2], alo[2];
    const _Float16* xrh = xh + (size_t)(i0 + hrow) * 64;
    const _Float16* xrl = xl + (size_t)(i0 + hrow) * 64;
    #pragma unroll
    for (int ks = 0; ks < 2; ++ks) {
        const int kb = ks * 32 + hsel * 16;
        ahi[ks].h[0] = *(const v8h*)(xrh + kb);
        ahi[ks].h[1] = *(const v8h*)(xrh + kb + 8);
        alo[ks].h[0] = *(const v8h*)(xrl + kb);
        alo[ks].h[1] = *(const v8h*)(xrl + kb + 8);
    }

    // bx for the 8 C/D rows this lane produces (row = v + 8*hsel)
    float bxr[8];
    #pragma unroll
    for (int v = 0; v < 8; ++v) bxr[v] = bx[i0 + v + 8 * hsel];

    const float sg = sigma[0];
    const float k2 = LOG2E / (sg * sg);

    for (int ct = 0; ct < 8; ++ct) {
        const int jc = j0 + ct * 16;
        // B uses the same contiguous k order as A: lane holds column jc+hrow,
        // K = hsel*16 .. hsel*16+15 (+32 per K-step) -> contiguous 32B.
        const _Float16* yrh = yh + (size_t)(jc + hrow) * 64 + hsel * 16;
        const _Float16* yrl = yl + (size_t)(jc + hrow) * 64 + hsel * 16;

        v8f acc = {};
        #pragma unroll
        for (int ks = 0; ks < 2; ++ks) {
            A16 bhi, blo;
            bhi.h[0] = *(const v8h*)(yrh + ks * 32);
            bhi.h[1] = *(const v8h*)(yrh + ks * 32 + 8);
            blo.h[0] = *(const v8h*)(yrl + ks * 32);
            blo.h[1] = *(const v8h*)(yrl + ks * 32 + 8);

            acc = __builtin_amdgcn_wmma_f32_16x16x32_f16(
                false, alo[ks].v, false, bhi.v, (short)0, acc, false, false);
            acc = __builtin_amdgcn_wmma_f32_16x16x32_f16(
                false, ahi[ks].v, false, blo.v, (short)0, acc, false, false);
            acc = __builtin_amdgcn_wmma_f32_16x16x32_f16(
                false, ahi[ks].v, false, bhi.v, (short)0, acc, false, false);
        }

        const float byv = by[jc + hrow];
        // C/D layout: VGPR v -> (row = v + 8*hsel, col = hrow)
        #pragma unroll
        for (int v = 0; v < 8; ++v) {
            float t = fmaf(acc[v], k2, bxr[v] + byv);
            t = fminf(t, 0.0f);                      // clamp sqd >= 0
            // native v_exp_f32 (TRANS pipe); t <= 0 so no overflow concern,
            // sub-2^-126 results flush toward 0 which is fine for an RBF tail.
            float r = __builtin_amdgcn_exp2f(t);
            __builtin_nontemporal_store(
                r, &out[(size_t)(i0 + v + 8 * hsel) * M + jc + hrow]);
        }
    }
}

// ---------------------------------------------------------------------------
extern "C" void kernel_launch(void* const* d_in, const int* in_sizes, int n_in,
                              void* d_out, int out_size, void* d_ws, size_t ws_size,
                              hipStream_t stream) {
    const float* x     = (const float*)d_in[0];
    const float* y     = (const float*)d_in[1];
    const float* sigma = (const float*)d_in[2];
    float* out = (float*)d_out;

    const int D = 64;
    const int N = in_sizes[0] / D;   // 8192
    const int M = in_sizes[1] / D;   // 8192

    // workspace layout (halves are 2B): xhi | xlo | yhi | ylo | bx | by
    _Float16* xhi = (_Float16*)d_ws;
    _Float16* xlo = xhi + (size_t)N * D;
    _Float16* yhi = xlo + (size_t)N * D;
    _Float16* ylo = yhi + (size_t)M * D;
    float*    bx  = (float*)(ylo + (size_t)M * D);
    float*    by  = bx + N;

    // prep: 8 waves (rows) per 256-thread block
    rbf_prep<<<(N + 7) / 8, 256, 0, stream>>>(x, xhi, xlo, bx, sigma, N);
    rbf_prep<<<(M + 7) / 8, 256, 0, stream>>>(y, yhi, ylo, by, sigma, M);

    // main: one wave per 16x128 output strip, 8 waves per block
    const int waves = (N / 16) * (M / 128);
    rbf_main<<<waves / 8, 256, 0, stream>>>(xhi, xlo, yhi, ylo, bx, by, sigma,
                                            out, M);
}
